// Attention_87668872446175
// MI455X (gfx1250) — matrix operand order repaired and checked by reading, not compiled, not run on hardware
//
#include <hip/hip_runtime.h>
#include <hip/hip_bf16.h>
#include <math.h>

// ---------------------------------------------------------------------------
// Fused multi-head attention fwd (b=8, n=1024, dim=768, h=12, dh=64) for
// gfx1250. All GEMMs via v_wmma_f32_16x16x32_bf16 (fp32 accumulate).
// GEMM B-operands staged per-workgroup through LDS via async-to-LDS
// (ASYNCcnt path) when the toolchain exposes the builtins.
// ---------------------------------------------------------------------------

typedef __bf16 bf16_t;
typedef __bf16 v16bf __attribute__((ext_vector_type(16)));
typedef __bf16 v8bf  __attribute__((ext_vector_type(8)));
typedef float  v8f   __attribute__((ext_vector_type(8)));
typedef float  v4f   __attribute__((ext_vector_type(4)));
typedef int    v4i   __attribute__((ext_vector_type(4)));

typedef __attribute__((address_space(1))) v4i gbl_v4i;  // global
typedef __attribute__((address_space(3))) v4i lds_v4i;  // LDS

#define HEADS 12
#define DHEAD 64
#define SEQ   1024
#define BATCH 8
#define DIM   768
#define INNER 768          // HEADS*DHEAD
#define NQKV  2304         // 3*INNER
#define ROWS  (BATCH*SEQ)  // 8192
#define AW    8            // waves per block

#if defined(__has_builtin)
#if __has_builtin(__builtin_amdgcn_global_load_async_to_lds_b128) && \
    __has_builtin(__builtin_amdgcn_s_wait_asynccnt)
#define HAVE_ASYNC_LDS 1
#endif
#endif
#ifndef HAVE_ASYNC_LDS
#define HAVE_ASYNC_LDS 0
#endif

static __device__ __forceinline__ v16bf cat8(v8bf a, v8bf b) {
  return __builtin_shufflevector(a, b, 0,1,2,3,4,5,6,7,8,9,10,11,12,13,14,15);
}

static __device__ __forceinline__ v8f wmma_bf16(v16bf a, v16bf b, v8f c) {
  // D = A(16x32 bf16) * B(32x16 bf16) + C(16x16 f32)
  return __builtin_amdgcn_wmma_f32_16x16x32_bf16(
      /*neg_a=*/false, a, /*neg_b=*/false, b,
      /*c_mod=*/(short)0, c, /*reuse_a=*/false, /*reuse_b=*/false);
}

// Stage 16 bytes global -> LDS (async engine if available).
static __device__ __forceinline__ void stage16(const bf16_t* g, bf16_t* l) {
#if HAVE_ASYNC_LDS
  __builtin_amdgcn_global_load_async_to_lds_b128(
      (gbl_v4i*)(void*)g, (lds_v4i*)(void*)l, 0, 0);
#else
  *(v8bf*)l = *(const v8bf*)g;
#endif
}

static __device__ __forceinline__ void stage_commit() {
#if HAVE_ASYNC_LDS
  __builtin_amdgcn_s_wait_asynccnt(0);
#endif
  __syncthreads();
}

// ----------------------------- converts ------------------------------------

__global__ void k_f32_to_bf16(const float* __restrict__ src,
                              bf16_t* __restrict__ dst, int n) {
  int i = blockIdx.x * blockDim.x + threadIdx.x;
  if (i < n) dst[i] = (bf16_t)src[i];
}

// dst[n*K + k] = (bf16) src[k*N + n]   (store W transposed: [N][K] row-major)
__global__ void k_transpose_bf16(const float* __restrict__ src,
                                 bf16_t* __restrict__ dst, int K, int N) {
  int i = blockIdx.x * blockDim.x + threadIdx.x;
  if (i >= K * N) return;
  int k = i % K;
  int n = i / K;
  dst[i] = (bf16_t)src[k * N + n];
}

// ----------------------------- QKV GEMM ------------------------------------
// Block = 8 waves, one 128-row x 64-col supertile of qkv = x @ w_qkv.
// B slab (64 cols x 32 K = 4KB) staged once per block per k-step into LDS
// (double buffered, async). Each wave owns a 16-row tile. Epilogue scatters
// into Q[b,h,n,64], K[b,h,n,64] (RoPE later) and V^T[b,h,64,n].

__global__ void __launch_bounds__(256) k_qkv_gemm(
    const bf16_t* __restrict__ xb,    // [8192][768] bf16 row-major
    const bf16_t* __restrict__ wT,    // [2304][768] bf16 (w_qkv transposed)
    bf16_t* __restrict__ qbuf, bf16_t* __restrict__ kbuf,
    bf16_t* __restrict__ vtbuf) {
  __shared__ __align__(32) bf16_t sB[2][64 * 32];   // 2 x 4KB

  const int tid  = threadIdx.x;
  const int lane = tid & 31;
  const int wv   = tid >> 5;                 // 0..7
  const int rowSup = ROWS / 128;             // 64
  const int rs = blockIdx.x % rowSup;
  const int cg = blockIdx.x / rowSup;        // 0..35
  if (cg >= NQKV / 64) return;
  const int i0 = rs * 128 + wv * 16;
  const int c0 = cg * 64;
  const int l15 = lane & 15, lhalf = lane >> 4;

  const bf16_t* arow = xb + (long)(i0 + l15) * DIM + lhalf * 16;
  // Staging: thread tid copies row (tid>>2) of the 64x32 B slab, 8 elements.
  const bf16_t* bsrc = wT + (long)(c0 + (tid >> 2)) * DIM + (tid & 3) * 8;
  bf16_t* bdst0 = &sB[0][tid * 8];
  bf16_t* bdst1 = &sB[1][tid * 8];

  stage16(bsrc, bdst0);
  stage_commit();

  v8f acc[4] = {};
  int cur = 0;
  for (int k0 = 0; k0 < DIM; k0 += 32) {
    if (k0 + 32 < DIM) stage16(bsrc + k0 + 32, cur ? bdst0 : bdst1);
    if (k0 + 64 < DIM) __builtin_prefetch(arow + k0 + 64, 0, 3);
    v16bf A = cat8(*(const v8bf*)(arow + k0), *(const v8bf*)(arow + k0 + 8));
    const bf16_t* lb = sB[cur];
#pragma unroll
    for (int t = 0; t < 4; ++t) {
      v16bf B = *(const v16bf*)(lb + (t * 16 + l15) * 32 + lhalf * 16);
      acc[t] = wmma_bf16(A, B, acc[t]);
    }
    stage_commit();
    cur ^= 1;
  }

  // Scatter epilogue: each 64-col group lies in exactly one (tensor, head).
  const int bb   = i0 >> 10;           // batch
  const int il   = i0 & (SEQ - 1);     // seq offset of tile
  const int tens = c0 / INNER;         // 0=q 1=k 2=v
  const int h    = (c0 % INNER) / DHEAD;
  const long bh  = (long)bb * HEADS + h;
#pragma unroll
  for (int t = 0; t < 4; ++t) {
    const int d = t * 16 + l15;
#pragma unroll
    for (int v = 0; v < 8; ++v) {
      const int m = v + 8 * lhalf;
      bf16_t val = (bf16_t)acc[t][v];
      if (tens == 0)
        qbuf[(bh * SEQ + il + m) * DHEAD + d] = val;
      else if (tens == 1)
        kbuf[(bh * SEQ + il + m) * DHEAD + d] = val;
      else
        vtbuf[(bh * DHEAD + d) * SEQ + il + m] = val;
    }
  }
}

// ------------------------------- RoPE --------------------------------------
// One thread per (tensor, b, h, pos, d<32) rotation pair, in place on bf16.

__global__ void k_rope(bf16_t* __restrict__ qbuf, bf16_t* __restrict__ kbuf) {
  const int perT = BATCH * HEADS * SEQ * 32;   // 3,145,728
  int tid = blockIdx.x * blockDim.x + threadIdx.x;
  if (tid >= 2 * perT) return;
  bf16_t* buf = (tid < perT) ? qbuf : kbuf;
  int t   = (tid < perT) ? tid : tid - perT;
  int d   = t & 31;
  int pos = (t >> 5) & (SEQ - 1);
  int bh  = t >> 15;
  long e  = ((long)bh * SEQ + pos) * DHEAD + d;
  // inv_freq = 10000^(-d/32) = exp(-d * ln(10000)/32)
  float inv = __expf(-(float)d * 0.28782314f);
  float th  = (float)pos * inv;
  float s, c;
  __sincosf(th, &s, &c);
  float x1 = (float)buf[e];
  float x2 = (float)buf[e + 32];
  buf[e]      = (bf16_t)(x1 * c - x2 * s);
  buf[e + 32] = (bf16_t)(x2 * c + x1 * s);
}

// --------------------------- Flash attention -------------------------------
// One wave per (b,h, 16-row query tile). j-loop in steps of 32:
//   S = Q*K^T (4 wmma), online softmax via LDS + half-wave shuffles,
//   O += P*V  (4 wmma into 16x64 fp32 accumulators).

__global__ void __launch_bounds__(256) k_attn(
    const bf16_t* __restrict__ qbuf, const bf16_t* __restrict__ kbuf,
    const bf16_t* __restrict__ vtbuf, bf16_t* __restrict__ obuf) {
  __shared__ __align__(16) float  sS[AW][16 * 32];
  __shared__ __align__(16) bf16_t sP[AW][16 * 32];
  __shared__ __align__(16) float  sRow[AW][16];

  const int lane   = threadIdx.x & 31;
  const int wlocal = threadIdx.x >> 5;
  const int wid    = blockIdx.x * AW + wlocal;
  const int itiles = SEQ / 16;               // 64
  const int bh = wid / itiles;               // 0..95
  const int it = wid % itiles;
  const int i0 = it * 16;
  const int l15 = lane & 15, lhalf = lane >> 4;

  const bf16_t* Q  = qbuf  + (long)bh * SEQ * DHEAD;
  const bf16_t* K  = kbuf  + (long)bh * SEQ * DHEAD;
  const bf16_t* VT = vtbuf + (long)bh * DHEAD * SEQ;

  float* myS  = sS[wlocal];
  bf16_t* myP = sP[wlocal];
  float* myR  = sRow[wlocal];

  // Q tile (16 rows x 64 d), as two A operands (d 0..31 and 32..63)
  const bf16_t* qrow = Q + (long)(i0 + l15) * DHEAD + lhalf * 16;
  v16bf AQ0 = cat8(*(const v8bf*)(qrow),      *(const v8bf*)(qrow + 8));
  v16bf AQ1 = cat8(*(const v8bf*)(qrow + 32), *(const v8bf*)(qrow + 40));

  v8f c0 = {}, c1 = {}, c2 = {}, c3 = {};
  float m_run = -3.0e38f, l_run = 0.f;
  const float scale = 0.125f;  // 1/sqrt(64)

  for (int j0 = 0; j0 < SEQ; j0 += 32) {
    // ---- S = Q K^T for 32 keys (two 16x16 tiles) ----
    v8f s0 = {}, s1 = {};
    {
      const bf16_t* kr0 = K + (long)(j0 + l15) * DHEAD + lhalf * 16;
      s0 = wmma_bf16(AQ0, *(const v16bf*)(kr0),      s0);
      s0 = wmma_bf16(AQ1, *(const v16bf*)(kr0 + 32), s0);
      const bf16_t* kr1 = K + (long)(j0 + 16 + l15) * DHEAD + lhalf * 16;
      s1 = wmma_bf16(AQ0, *(const v16bf*)(kr1),      s1);
      s1 = wmma_bf16(AQ1, *(const v16bf*)(kr1 + 32), s1);
    }
    // stage scaled scores to LDS row-major [16][32]  (same-wave DS in-order)
#pragma unroll
    for (int v = 0; v < 8; ++v) {
      int m = v + 8 * lhalf;
      myS[m * 32 + l15]      = s0[v] * scale;
      myS[m * 32 + 16 + l15] = s1[v] * scale;
    }
    // ---- online softmax: lane handles row l15, cols lhalf*16..+15 ----
    float vals[16];
    float pmax = -3.0e38f;
#pragma unroll
    for (int c = 0; c < 16; ++c) {
      float vv = myS[l15 * 32 + lhalf * 16 + c];
      vals[c] = vv;
      pmax = fmaxf(pmax, vv);
    }
    float rmax  = fmaxf(pmax, __shfl_xor(pmax, 16, 32));
    float m_new = fmaxf(m_run, rmax);
    float psum = 0.f;
#pragma unroll
    for (int c = 0; c < 16; ++c) {
      float p = __expf(vals[c] - m_new);
      psum += p;
      myP[l15 * 32 + lhalf * 16 + c] = (bf16_t)p;
    }
    float rsum  = psum + __shfl_xor(psum, 16, 32);
    float alpha = __expf(m_run - m_new);
    l_run = alpha * l_run + rsum;
    m_run = m_new;
    if (lhalf == 0) myR[l15] = alpha;
    // rescale accumulators by per-row alpha (row-mapped: row = v + 8*lhalf)
    v4f a0 = *(const v4f*)(myR + lhalf * 8);
    v4f a1 = *(const v4f*)(myR + lhalf * 8 + 4);
    float av[8] = {a0[0], a0[1], a0[2], a0[3], a1[0], a1[1], a1[2], a1[3]};
#pragma unroll
    for (int v = 0; v < 8; ++v) {
      c0[v] *= av[v]; c1[v] *= av[v]; c2[v] *= av[v]; c3[v] *= av[v];
    }
    // ---- O += P * V : P as A (16x32 bf16 from LDS), V^T rows as B ----
    const bf16_t* prow = myP + l15 * 32 + lhalf * 16;
    v16bf AP = cat8(*(const v8bf*)prow, *(const v8bf*)(prow + 8));
    const bf16_t* vb = VT + (long)l15 * SEQ + j0 + lhalf * 16;
    c0 = wmma_bf16(AP, *(const v16bf*)(vb),            c0);
    c1 = wmma_bf16(AP, *(const v16bf*)(vb + 16 * SEQ), c1);
    c2 = wmma_bf16(AP, *(const v16bf*)(vb + 32 * SEQ), c2);
    c3 = wmma_bf16(AP, *(const v16bf*)(vb + 48 * SEQ), c3);
  }

  // final 1/l normalization, write O as bf16 [8192][768]
  if (lhalf == 0) myR[l15] = 1.0f / l_run;
  v4f r0 = *(const v4f*)(myR + lhalf * 8);
  v4f r1 = *(const v4f*)(myR + lhalf * 8 + 4);
  float rv[8] = {r0[0], r0[1], r0[2], r0[3], r1[0], r1[1], r1[2], r1[3]};
  const int bb = bh / HEADS, h = bh % HEADS;
  bf16_t* orow = obuf + ((long)bb * SEQ + i0) * INNER + h * DHEAD;
#pragma unroll
  for (int v = 0; v < 8; ++v) {
    int m = v + 8 * lhalf;
    bf16_t* op = orow + (long)m * INNER;
    op[0 * 16 + l15] = (bf16_t)(c0[v] * rv[v]);
    op[1 * 16 + l15] = (bf16_t)(c1[v] * rv[v]);
    op[2 * 16 + l15] = (bf16_t)(c2[v] * rv[v]);
    op[3 * 16 + l15] = (bf16_t)(c3[v] * rv[v]);
  }
}

// ---------------------------- output GEMM ----------------------------------
// Same block structure as QKV GEMM: 128 rows x 64 cols per block, B slab
// staged through LDS (double buffered, async).

__global__ void __launch_bounds__(256) k_out_gemm(
    const bf16_t* __restrict__ ob,    // [8192][768] bf16
    const bf16_t* __restrict__ wT,    // [768][768] bf16 (w_out transposed)
    const float* __restrict__ bias, float* __restrict__ out) {
  __shared__ __align__(32) bf16_t sB[2][64 * 32];

  const int tid  = threadIdx.x;
  const int lane = tid & 31;
  const int wv   = tid >> 5;
  const int rowSup = ROWS / 128;       // 64
  const int rs = blockIdx.x % rowSup;
  const int cg = blockIdx.x / rowSup;  // 0..11
  if (cg >= DIM / 64) return;
  const int i0 = rs * 128 + wv * 16;
  const int c0 = cg * 64;
  const int l15 = lane & 15, lhalf = lane >> 4;

  const bf16_t* arow = ob + (long)(i0 + l15) * INNER + lhalf * 16;
  const bf16_t* bsrc = wT + (long)(c0 + (tid >> 2)) * INNER + (tid & 3) * 8;
  bf16_t* bdst0 = &sB[0][tid * 8];
  bf16_t* bdst1 = &sB[1][tid * 8];

  stage16(bsrc, bdst0);
  stage_commit();

  v8f acc[4] = {};
  int cur = 0;
  for (int k0 = 0; k0 < INNER; k0 += 32) {
    if (k0 + 32 < INNER) stage16(bsrc + k0 + 32, cur ? bdst0 : bdst1);
    if (k0 + 64 < INNER) __builtin_prefetch(arow + k0 + 64, 0, 3);
    v16bf A = cat8(*(const v8bf*)(arow + k0), *(const v8bf*)(arow + k0 + 8));
    const bf16_t* lb = sB[cur];
#pragma unroll
    for (int t = 0; t < 4; ++t) {
      v16bf B = *(const v16bf*)(lb + (t * 16 + l15) * 32 + lhalf * 16);
      acc[t] = wmma_bf16(A, B, acc[t]);
    }
    stage_commit();
    cur ^= 1;
  }

#pragma unroll
  for (int t = 0; t < 4; ++t) {
    const int col = c0 + t * 16 + l15;
    const float bv = bias[col];
#pragma unroll
    for (int v = 0; v < 8; ++v) {
      int m = v + 8 * lhalf;
      out[(long)(i0 + m) * DIM + col] = acc[t][v] + bv;
    }
  }
}

// ------------------------------ launcher -----------------------------------

extern "C" void kernel_launch(void* const* d_in, const int* in_sizes, int n_in,
                              void* d_out, int out_size, void* d_ws,
                              size_t ws_size, hipStream_t stream) {
  const float* x     = (const float*)d_in[0];  // [8,1024,768]
  const float* w_qkv = (const float*)d_in[1];  // [768,2304]
  const float* w_out = (const float*)d_in[2];  // [768,768]
  const float* b_out = (const float*)d_in[3];  // [768]
  float* out = (float*)d_out;                  // [8,1024,768]

  char* ws = (char*)d_ws;
  size_t off = 0;
  auto alloc = [&](size_t bytes) -> void* {
    void* p = ws + off;
    off = (off + bytes + 255) & ~(size_t)255;
    return p;
  };
  bf16_t* xb    = (bf16_t*)alloc((size_t)ROWS * DIM * 2);                  // 12.6 MB
  bf16_t* wqkvT = (bf16_t*)alloc((size_t)NQKV * DIM * 2);                  //  3.5 MB
  bf16_t* woutT = (bf16_t*)alloc((size_t)DIM * INNER * 2);                 //  1.2 MB
  bf16_t* qb    = (bf16_t*)alloc((size_t)BATCH * HEADS * SEQ * DHEAD * 2); // 12.6 MB
  bf16_t* kb    = (bf16_t*)alloc((size_t)BATCH * HEADS * SEQ * DHEAD * 2); // 12.6 MB
  bf16_t* vt    = (bf16_t*)alloc((size_t)BATCH * HEADS * DHEAD * SEQ * 2); // 12.6 MB
  bf16_t* obuf  = (bf16_t*)alloc((size_t)ROWS * INNER * 2);                // 12.6 MB

  {
    int n = ROWS * DIM;
    k_f32_to_bf16<<<(n + 255) / 256, 256, 0, stream>>>(x, xb, n);
  }
  k_transpose_bf16<<<(NQKV * DIM + 255) / 256, 256, 0, stream>>>(
      w_qkv, wqkvT, DIM, NQKV);
  k_transpose_bf16<<<(DIM * INNER + 255) / 256, 256, 0, stream>>>(
      w_out, woutT, INNER, DIM);

  // 64 row-supertiles (128 rows) * 36 col-groups
  k_qkv_gemm<<<64 * 36, 256, 0, stream>>>(xb, wqkvT, qb, kb, vt);

  {
    int n = 2 * BATCH * HEADS * SEQ * 32;
    k_rope<<<(n + 255) / 256, 256, 0, stream>>>(qb, kb);
  }

  // 96 (b,h) * 64 i-tiles = 6144 waves, 8 waves/block
  k_attn<<<(BATCH * HEADS * (SEQ / 16)) / AW, 256, 0, stream>>>(qb, kb, vt,
                                                                obuf);

  // 64 row-supertiles * 12 col-groups
  k_out_gemm<<<64 * 12, 256, 0, stream>>>(obuf, woutT, b_out, out);
}
